// BiLSTM_CRF_8641474200060
// MI455X (gfx1250) — matrix operand order repaired and checked by reading, not compiled, not run on hardware
//
#include <hip/hip_runtime.h>
#include <hip/hip_bf16.h>

// ---------------------------------------------------------------------------
// BiLSTM-CRF forward NLL on gfx1250 (MI455X), wave32 + WMMA f16->f32.
//  - xg kept in f16 (halves HBM traffic on the serial LSTM path)
//  - global_prefetch of next-step xg slab overlaps the recurrence's WMMA work
// ---------------------------------------------------------------------------

#define T_LEN 512
#define BATCH 128
#define EMB   256
#define HID   512
#define Hh    256          // H = HID/2
#define G4    1024         // 4*H
#define NC    25
#define PADT  1

typedef _Float16 half_t;
typedef __attribute__((ext_vector_type(16))) _Float16 v16h;
typedef __attribute__((ext_vector_type(8)))  float    v8f;

// ---------------- WMMA fragment helpers (ISA 7.12.2 layouts) ----------------
// A (16x32, f16, row-major src): lane 0-15 -> M=lane, K in {0..7,16..23};
//                                lane 16-31 -> M=lane-16, K in {8..15,24..31}
__device__ __forceinline__ v16h load_a_frag(const half_t* A, int ld, int m0, int k0, int lane) {
  v16h a;
  const int m = m0 + (lane & 15);
  const int kbase = (lane & 16) ? 8 : 0;
  const half_t* p = A + (size_t)m * ld + k0 + kbase;
#pragma unroll
  for (int j = 0; j < 16; ++j) {
    const int kk = (j < 8) ? j : (j + 8);
    a[j] = p[kk];
  }
  return a;
}

// B (32x16, f16) with element (k, n) = W[n*ld + k] (weights are row-major (N,K)):
// lane 0-15 -> N=lane, K in {0..7,16..23} ; lane 16-31 -> N=lane-16, K in {8..15,24..31}
__device__ __forceinline__ v16h load_b_frag(const half_t* W, int ld, int n0, int k0, int lane) {
  v16h b;
  const int n = n0 + (lane & 15);
  const int kbase = (lane & 16) ? 8 : 0;
  const half_t* p = W + (size_t)n * ld + k0 + kbase;
#pragma unroll
  for (int j = 0; j < 16; ++j) b[j] = p[(j < 8) ? j : (j + 8)];
  return b;
}

__device__ __forceinline__ v8f wmma32(v16h a, v16h b, v8f c) {
  return __builtin_amdgcn_wmma_f32_16x16x32_f16(false, a, false, b, (short)0, c, false, false);
}

__device__ __forceinline__ float sigm(float x) { return 1.f / (1.f + __expf(-x)); }

// ---------------------------- prep kernels ----------------------------------
__global__ void embed_cast_kernel(const int* __restrict__ sent,
                                  const float* __restrict__ emb,
                                  half_t* __restrict__ xh) {
  const size_t tid = (size_t)blockIdx.x * blockDim.x + threadIdx.x; // T*B*EMB
  if (tid >= (size_t)T_LEN * BATCH * EMB) return;
  const int e = (int)(tid & (EMB - 1));
  const size_t tb = tid >> 8; // / EMB
  const int tok = sent[tb];
  xh[tid] = (half_t)emb[(size_t)tok * EMB + e];
}

__global__ void cast_h_kernel(const float* __restrict__ src, half_t* __restrict__ dst, int n) {
  const int i = blockIdx.x * blockDim.x + threadIdx.x;
  if (i < n) dst[i] = (half_t)src[i];
}

__global__ void pad_we_kernel(const float* __restrict__ we, half_t* __restrict__ weh) {
  const int i = blockIdx.x * blockDim.x + threadIdx.x; // 32*512
  if (i >= 32 * HID) return;
  const int n = i >> 9, k = i & (HID - 1);
  weh[i] = (n < NC) ? (half_t)we[n * HID + k] : (half_t)0.f;
}

// ------- xg = x * W_ih^T + b_ih + b_hh  (M=65536,K=256,N=1024), f16 out -----
__global__ void gemm_xg_kernel(const half_t* __restrict__ xh,
                               const half_t* __restrict__ wih,
                               const float* __restrict__ bih,
                               const float* __restrict__ bhh,
                               half_t* __restrict__ xg) {
  const int lane = threadIdx.x & 31;
  const int wv = blockIdx.x * (blockDim.x >> 5) + (threadIdx.x >> 5);
  const int mt = wv >> 6;          // 4096 row tiles
  const int nt = wv & 63;          // 64 col tiles
  const int m0 = mt * 16, n0 = nt * 16;
  v8f acc = {};
#pragma unroll
  for (int k0 = 0; k0 < EMB; k0 += 32) {
    v16h a = load_a_frag(xh, EMB, m0, k0, lane);
    v16h b = load_b_frag(wih, EMB, n0, k0, lane);
    acc = wmma32(a, b, acc);
  }
  const int n = n0 + (lane & 15);
  const float bias = bih[n] + bhh[n];
  const int mo = (lane & 16) ? 8 : 0;
#pragma unroll
  for (int r = 0; r < 8; ++r) {
    const int m = m0 + mo + r;
    xg[(size_t)m * G4 + n] = (half_t)(acc[r] + bias);
  }
}

// ------------- persistent bidirectional LSTM (1 block per direction) --------
// LDS: h double-buffered f16 (2*64KB) + c f32 (128KB) = 256KB (CDNA5 320KB WGP)
__global__ void __launch_bounds__(1024) lstm_kernel(const half_t* __restrict__ xg_f,
                                                    const half_t* __restrict__ xg_b,
                                                    const half_t* __restrict__ whh_f,
                                                    const half_t* __restrict__ whh_b,
                                                    half_t* __restrict__ hs) {
  const int dir = blockIdx.x;                       // 0=fwd, 1=bwd
  const half_t* xg = dir ? xg_b : xg_f;
  const half_t* whh = dir ? whh_b : whh_f;
  const int hofs = dir ? Hh : 0;

  extern __shared__ char smem[];
  half_t* hb0 = (half_t*)smem;                      // 128*256 f16
  half_t* hb1 = hb0 + BATCH * Hh;
  float*  cb  = (float*)(hb1 + BATCH * Hh);         // 128*256 f32

  const int tid = threadIdx.x;                      // 1024 threads = 32 waves
  const int lane = tid & 31;
  const int wv = tid >> 5;

  for (int i = tid; i < BATCH * Hh; i += 1024) { hb0[i] = (half_t)0.f; cb[i] = 0.f; }
  __syncthreads();

  for (int s = 0; s < T_LEN; ++s) {
    const int t = dir ? (T_LEN - 1 - s) : s;
    half_t* hcur = (s & 1) ? hb1 : hb0;
    half_t* hnxt = (s & 1) ? hb0 : hb1;

    // Pull next timestep's 256KB xg slab toward this WGP while we compute.
    if (s + 1 < T_LEN) {
      const int tn = dir ? (T_LEN - 2 - s) : (s + 1);
      const char* base = (const char*)(xg + (size_t)tn * BATCH * G4);
#pragma unroll
      for (int i = 0; i < 2; ++i) {
        __builtin_prefetch(base + ((size_t)(tid + i * 1024)) * 128, 0, 3);
      }
    }

    for (int ti = 0; ti < 4; ++ti) {
      const int task = wv * 4 + ti;                 // 0..127
      const int mt = task & 7;                      // batch tile (128/16)
      const int nt = task >> 3;                     // hidden tile (256/16)
      const int b0 = mt * 16, h0 = nt * 16;

      v8f acc[4] = {};
      for (int k0 = 0; k0 < Hh; k0 += 32) {
        v16h a = load_a_frag(hcur, Hh, b0, k0, lane);   // from LDS
#pragma unroll
        for (int q = 0; q < 4; ++q) {
          v16h bf = load_b_frag(whh, Hh, q * Hh + h0, k0, lane);
          acc[q] = wmma32(a, bf, acc[q]);
        }
      }

      const int hcol = h0 + (lane & 15);
      const int mo = (lane & 16) ? 8 : 0;
#pragma unroll
      for (int r = 0; r < 8; ++r) {
        const int bi = b0 + mo + r;
        const size_t row = ((size_t)t * BATCH + bi) * G4;
        const float gi = acc[0][r] + (float)xg[row + 0 * Hh + hcol];
        const float gf = acc[1][r] + (float)xg[row + 1 * Hh + hcol];
        const float gg = acc[2][r] + (float)xg[row + 2 * Hh + hcol];
        const float go = acc[3][r] + (float)xg[row + 3 * Hh + hcol];
        const float iv = sigm(gi), fv = sigm(gf), ov = sigm(go);
        const float cv = fv * cb[bi * Hh + hcol] + iv * tanhf(gg);
        const float hv = ov * tanhf(cv);
        cb[bi * Hh + hcol] = cv;
        hnxt[bi * Hh + hcol] = (half_t)hv;
        hs[((size_t)t * BATCH + bi) * HID + hofs + hcol] = (half_t)hv;
      }
    }
    __syncthreads();
  }
}

// -------- emissions = hs * W_e^T + b_e  (M=65536,K=512,N=25 padded to 32) ---
__global__ void gemm_emis_kernel(const half_t* __restrict__ hs,
                                 const half_t* __restrict__ weh,
                                 const float* __restrict__ be,
                                 float* __restrict__ emis) {
  const int lane = threadIdx.x & 31;
  const int wv = blockIdx.x * (blockDim.x >> 5) + (threadIdx.x >> 5);
  const int mt = wv >> 1, nt = wv & 1;
  const int m0 = mt * 16, n0 = nt * 16;
  v8f acc = {};
#pragma unroll
  for (int k0 = 0; k0 < HID; k0 += 32) {
    v16h a = load_a_frag(hs, HID, m0, k0, lane);
    v16h b = load_b_frag(weh, HID, n0, k0, lane);
    acc = wmma32(a, b, acc);
  }
  const int n = n0 + (lane & 15);
  if (n < NC) {
    const float bias = be[n];
    const int mo = (lane & 16) ? 8 : 0;
#pragma unroll
    for (int r = 0; r < 8; ++r) {
      const int m = m0 + mo + r;
      emis[(size_t)m * NC + n] = acc[r] + bias;
    }
  }
}

// --------------------------- CRF NLL (per batch) ----------------------------
__global__ void crf_kernel(const int* __restrict__ sent,
                           const int* __restrict__ tagsp,
                           const float* __restrict__ emis,
                           const float* __restrict__ st,
                           const float* __restrict__ et,
                           const float* __restrict__ trans,
                           float* __restrict__ out) {
  const int b = blockIdx.x;        // 128 blocks
  const int lane = threadIdx.x;    // 32 threads = 1 wave
  __shared__ float trans_s[NC * NC];
  __shared__ float score[NC];
  __shared__ float red[32];
  __shared__ int   redc[32];

  for (int i = lane; i < NC * NC; i += 32) trans_s[i] = trans[i];
  __syncthreads();

  // ---- numerator (gold path score) ----
  float part = 0.f;
  int cnt = 0;
  for (int t = lane; t < T_LEN; t += 32) {
    const int m = (sent[t * BATCH + b] != PADT) ? 1 : 0;
    cnt += m;
    const int tg = tagsp[t * BATCH + b];
    const float e = emis[((size_t)t * BATCH + b) * NC + tg];
    if (t == 0) {
      part += st[tg] + e;
    } else if (m) {
      const int tgp = tagsp[(t - 1) * BATCH + b];
      part += trans_s[tgp * NC + tg] + e;
    }
  }
  red[lane] = part;
  redc[lane] = cnt;
  __syncthreads();
  if (lane == 0) {
    float s = 0.f; int total = 0;
    for (int i = 0; i < 32; ++i) { s += red[i]; total += redc[i]; }
    const int lt = tagsp[(total - 1) * BATCH + b];
    red[0] = s + et[lt];
  }
  __syncthreads();
  const float num = red[0];

  // ---- partition function (forward algorithm) ----
  if (lane < NC) score[lane] = st[lane] + emis[(size_t)b * NC + lane];
  __syncthreads();
  for (int t = 1; t < T_LEN; ++t) {
    const int m = (sent[t * BATCH + b] != PADT);
    float v = 0.f;
    if (lane < NC) {
      float mx = -INFINITY;
      for (int i = 0; i < NC; ++i)
        mx = fmaxf(mx, score[i] + trans_s[i * NC + lane]);
      float sum = 0.f;
      for (int i = 0; i < NC; ++i)
        sum += __expf(score[i] + trans_s[i * NC + lane] - mx);
      v = mx + __logf(sum) + emis[((size_t)t * BATCH + b) * NC + lane];
    }
    __syncthreads();
    if (m && lane < NC) score[lane] = v;
    __syncthreads();
  }
  if (lane == 0) {
    float mx = -INFINITY;
    for (int i = 0; i < NC; ++i) mx = fmaxf(mx, score[i] + et[i]);
    float sum = 0.f;
    for (int i = 0; i < NC; ++i) sum += __expf(score[i] + et[i] - mx);
    const float denom = mx + __logf(sum);
    atomicAdd(out, denom - num);   // accumulate -llh
  }
}

// ------------------------------- launcher -----------------------------------
extern "C" void kernel_launch(void* const* d_in, const int* in_sizes, int n_in,
                              void* d_out, int out_size, void* d_ws, size_t ws_size,
                              hipStream_t stream) {
  const int*   sentence = (const int*)d_in[0];
  const int*   tags     = (const int*)d_in[1];
  const float* emb      = (const float*)d_in[2];
  const float* w_ih_f   = (const float*)d_in[3];
  const float* w_hh_f   = (const float*)d_in[4];
  const float* b_ih_f   = (const float*)d_in[5];
  const float* b_hh_f   = (const float*)d_in[6];
  const float* w_ih_b   = (const float*)d_in[7];
  const float* w_hh_b   = (const float*)d_in[8];
  const float* b_ih_b   = (const float*)d_in[9];
  const float* b_hh_b   = (const float*)d_in[10];
  const float* W_e      = (const float*)d_in[11];
  const float* b_e      = (const float*)d_in[12];
  const float* start_tr = (const float*)d_in[13];
  const float* end_tr   = (const float*)d_in[14];
  const float* trans    = (const float*)d_in[15];
  float* out = (float*)d_out;

  // workspace carve
  size_t off = 0;
  auto carve = [&](size_t bytes) -> void* {
    void* p = (char*)d_ws + off;
    off += (bytes + 255) & ~(size_t)255;
    return p;
  };
  half_t* x_h      = (half_t*)carve((size_t)T_LEN * BATCH * EMB * 2);
  half_t* w_ih_f_h = (half_t*)carve((size_t)G4 * EMB * 2);
  half_t* w_hh_f_h = (half_t*)carve((size_t)G4 * Hh * 2);
  half_t* w_ih_b_h = (half_t*)carve((size_t)G4 * EMB * 2);
  half_t* w_hh_b_h = (half_t*)carve((size_t)G4 * Hh * 2);
  half_t* w_e_h    = (half_t*)carve((size_t)32 * HID * 2);
  half_t* xg_f     = (half_t*)carve((size_t)T_LEN * BATCH * G4 * 2);
  half_t* xg_b     = (half_t*)carve((size_t)T_LEN * BATCH * G4 * 2);
  half_t* hs_h     = (half_t*)carve((size_t)T_LEN * BATCH * HID * 2);
  float*  emis     = (float*)carve((size_t)T_LEN * BATCH * NC * 4);
  (void)ws_size; (void)in_sizes; (void)n_in; (void)out_size;

  hipMemsetAsync(d_out, 0, sizeof(float), stream);

  // 1) embedding gather + f16 cast
  {
    const size_t total = (size_t)T_LEN * BATCH * EMB;
    embed_cast_kernel<<<(unsigned)((total + 255) / 256), 256, 0, stream>>>(sentence, emb, x_h);
  }
  // 2) weight casts
  {
    const int nw = G4 * EMB;  // == G4*Hh == 262144
    cast_h_kernel<<<(nw + 255) / 256, 256, 0, stream>>>(w_ih_f, w_ih_f_h, nw);
    cast_h_kernel<<<(nw + 255) / 256, 256, 0, stream>>>(w_hh_f, w_hh_f_h, nw);
    cast_h_kernel<<<(nw + 255) / 256, 256, 0, stream>>>(w_ih_b, w_ih_b_h, nw);
    cast_h_kernel<<<(nw + 255) / 256, 256, 0, stream>>>(w_hh_b, w_hh_b_h, nw);
    pad_we_kernel<<<(32 * HID + 255) / 256, 256, 0, stream>>>(W_e, w_e_h);
  }
  // 3) input-projection GEMMs (WMMA): 4096x64 tiles, 8 waves/block
  {
    const int tiles = (T_LEN * BATCH / 16) * (G4 / 16);
    const int blocks = tiles / 8;
    gemm_xg_kernel<<<blocks, 256, 0, stream>>>(x_h, w_ih_f_h, b_ih_f, b_hh_f, xg_f);
    gemm_xg_kernel<<<blocks, 256, 0, stream>>>(x_h, w_ih_b_h, b_ih_b, b_hh_b, xg_b);
  }
  // 4) persistent LSTM, one block per direction, 256KB LDS
  {
    const size_t lds = (size_t)2 * BATCH * Hh * sizeof(half_t) + (size_t)BATCH * Hh * sizeof(float);
    lstm_kernel<<<2, 1024, lds, stream>>>(xg_f, xg_b, w_hh_f_h, w_hh_b_h, hs_h);
  }
  // 5) emission GEMM (WMMA), 4096x2 tiles
  {
    const int tiles = (T_LEN * BATCH / 16) * 2;
    gemm_emis_kernel<<<tiles / 8, 256, 0, stream>>>(hs_h, w_e_h, b_e, emis);
  }
  // 6) CRF NLL
  crf_kernel<<<BATCH, 32, 0, stream>>>(sentence, tags, emis, start_tr, end_tr, trans, out);
}
